// LSTMModel_2929167696643
// MI455X (gfx1250) — matrix operand order repaired and checked by reading, not compiled, not run on hardware
//
#include <hip/hip_runtime.h>
#include <hip/hip_bf16.h>

#define Bsz  128
#define Tlen 1024
#define Din  256
#define Hdim 1024
#define Odim 10
#define G4   (4 * Hdim)

typedef __bf16 bf16;
typedef __attribute__((ext_vector_type(16))) __bf16 v16bf;
typedef __attribute__((ext_vector_type(8)))  __bf16 v8bf;
typedef __attribute__((ext_vector_type(8)))  float  v8f;

__device__ __forceinline__ bf16 f2bf(float f) {
    union { float f; unsigned u; } v; v.f = f;
    unsigned r = v.u + 0x7FFFu + ((v.u >> 16) & 1u);   // RNE truncate to bf16
    unsigned short hs = (unsigned short)(r >> 16);
    bf16 out;
    __builtin_memcpy(&out, &hs, 2);
    return out;
}

__device__ __forceinline__ float sigmoidf_fast(float x) {
    return 1.0f / (1.0f + __expf(-x));
}

// A-fragment (16x32 bf16) from a bf16 row: two contiguous 16B loads, no ALU.
// lane<16: K = k0+0..15 ; lane>=16: K = k0+16..31 (contiguous, matches B)
__device__ __forceinline__ v16bf lda_bf(const bf16* base, int half) {
    v8bf lo = *(const v8bf*)(base + half * 16);
    v8bf hi = *(const v8bf*)(base + half * 16 + 8);
    return __builtin_shufflevector(lo, hi, 0,1,2,3,4,5,6,7,8,9,10,11,12,13,14,15);
}

// A-fragment from an f32 row (fallback path): load + RNE convert.
__device__ __forceinline__ v16bf lda_f32(const float* base, int half) {
    v8f lo = *(const v8f*)(base + half * 16);
    v8f hi = *(const v8f*)(base + half * 16 + 8);
    v16bf a;
#pragma unroll
    for (int j = 0; j < 8; ++j) { a[j] = f2bf(lo[j]); a[j + 8] = f2bf(hi[j]); }
    return a;
}

// -------- prep: Wx[D][4H] f32 -> WxT[4H][D] bf16 (transpose + convert) --------
__global__ void prep_wx(const float* __restrict__ Wx, bf16* __restrict__ WxT) {
    int idx = blockIdx.x * blockDim.x + threadIdx.x;
    if (idx >= G4 * Din) return;
    int n = idx / Din;
    int d = idx - n * Din;
    WxT[idx] = f2bf(Wx[(size_t)d * G4 + n]);
}

// -------- prep: Wh[H][4H] f32 -> WhT[4H][H] bf16 --------
__global__ void prep_wh(const float* __restrict__ Wh, bf16* __restrict__ WhT) {
    int idx = blockIdx.x * blockDim.x + threadIdx.x;
    if (idx >= G4 * Hdim) return;
    int n = idx / Hdim;
    int k = idx - n * Hdim;
    WhT[idx] = f2bf(Wh[(size_t)k * G4 + n]);
}

// -------- prep: x f32 -> bf16 (same [B,T,D] layout) --------
__global__ void prep_x(const float* __restrict__ x, bf16* __restrict__ xb, int n) {
    int idx = blockIdx.x * blockDim.x + threadIdx.x;
    if (idx >= n) return;
    xb[idx] = f2bf(x[idx]);
}

// -------- zero bf16 h0 and f32 c --------
__global__ void zero_state(bf16* __restrict__ h0, float* __restrict__ c) {
    int idx = blockIdx.x * blockDim.x + threadIdx.x;
    if (idx >= Bsz * Hdim) return;
    h0[idx] = f2bf(0.0f);
    c[idx]  = 0.0f;
}

// -------- one LSTM step: gates = x_t@Wx + h@Wh + b; update c, h --------
// 64 blocks x 128 threads = 256 waves; wave -> 32 batch rows x 16 hidden cols,
// all 4 gates (8 f32 accumulators). Waves in a block share nTile -> same B tiles.
template <bool XBF>
__global__ __launch_bounds__(128)
void lstm_step(const float* __restrict__ xf,
               const bf16*  __restrict__ xb,
               const bf16*  __restrict__ WxT,
               const bf16*  __restrict__ WhT,
               const float* __restrict__ bias,
               const bf16*  __restrict__ h_in,
               bf16*        __restrict__ h_out,
               float*       __restrict__ h_f32,
               float*       __restrict__ c,
               int t)
{
    const int lane   = threadIdx.x & 31;
    const int wave   = blockIdx.x * (blockDim.x >> 5) + (threadIdx.x >> 5);
    const int mSuper = wave & 3;    // 4 super-tiles of 32 batch rows
    const int nTile  = wave >> 2;   // 64 hidden tiles
    const int m0 = mSuper * 32;
    const int n0 = nTile * 16;
    const int mrow = lane & 15;
    const int half = lane >> 4;
    const int ncol = lane & 15;

    v8f acc[8] = {};  // [sub-tile(2)][gate(4)] accumulators

    // ---- K-part 1: x_t (K = 0..D) ----
    const size_t xoff0 = ((size_t)(m0 + mrow) * Tlen + (size_t)t) * Din;
    const size_t xoff1 = ((size_t)(m0 + 16 + mrow) * Tlen + (size_t)t) * Din;
    for (int k0 = 0; k0 < Din; k0 += 32) {
        v16bf a0, a1;
        if (XBF) {
            a0 = lda_bf(xb + xoff0 + k0, half);
            a1 = lda_bf(xb + xoff1 + k0, half);
        } else {
            a0 = lda_f32(xf + xoff0 + k0, half);
            a1 = lda_f32(xf + xoff1 + k0, half);
        }
#pragma unroll
        for (int g = 0; g < 4; ++g) {
            const bf16* bp = WxT + (size_t)(g * Hdim + n0 + ncol) * Din + k0 + half * 16;
            v16bf bfrag = *(const v16bf*)bp;
            acc[g]     = __builtin_amdgcn_wmma_f32_16x16x32_bf16(
                false, a0, false, bfrag, (short)0, acc[g], false, false);
            acc[4 + g] = __builtin_amdgcn_wmma_f32_16x16x32_bf16(
                false, a1, false, bfrag, (short)0, acc[4 + g], false, false);
        }
    }

    // ---- K-part 2: h_prev (K = 0..H), bf16 A fragments, zero conversion ALU ----
    const bf16* h0p = h_in + (size_t)(m0 + mrow) * Hdim;
    const bf16* h1p = h_in + (size_t)(m0 + 16 + mrow) * Hdim;
    for (int k0 = 0; k0 < Hdim; k0 += 32) {
        v16bf a0 = lda_bf(h0p + k0, half);
        v16bf a1 = lda_bf(h1p + k0, half);
#pragma unroll
        for (int g = 0; g < 4; ++g) {
            const bf16* bp = WhT + (size_t)(g * Hdim + n0 + ncol) * Hdim + k0 + half * 16;
            v16bf bfrag = *(const v16bf*)bp;
            acc[g]     = __builtin_amdgcn_wmma_f32_16x16x32_bf16(
                false, a0, false, bfrag, (short)0, acc[g], false, false);
            acc[4 + g] = __builtin_amdgcn_wmma_f32_16x16x32_bf16(
                false, a1, false, bfrag, (short)0, acc[4 + g], false, false);
        }
    }

    // ---- epilogue: bias + activations + c/h update ----
    const int ng = n0 + ncol;
    const float bi = bias[0 * Hdim + ng];
    const float bf_ = bias[1 * Hdim + ng];
    const float bg = bias[2 * Hdim + ng];
    const float bo = bias[3 * Hdim + ng];

#pragma unroll
    for (int s = 0; s < 2; ++s) {
#pragma unroll
        for (int r = 0; r < 8; ++r) {
            const int m = m0 + s * 16 + r + half * 8;  // C/D row: M = r + 8*(lane/16)
            const size_t off = (size_t)m * Hdim + ng;
            const float iv = sigmoidf_fast(acc[s * 4 + 0][r] + bi);
            const float fv = sigmoidf_fast(acc[s * 4 + 1][r] + bf_);
            const float gv = tanhf(acc[s * 4 + 2][r] + bg);
            const float ov = sigmoidf_fast(acc[s * 4 + 3][r] + bo);
            const float cn = fv * c[off] + iv * gv;
            c[off] = cn;
            const float hv = ov * tanhf(cn);
            h_f32[off] = hv;        // for final decode (full f32)
            h_out[off] = f2bf(hv);  // for next step's A fragments
        }
    }
}

// -------- final decode: out[b][o] = h_T[b]·Wd[:,o] + bd[o] --------
__global__ void decode(const float* __restrict__ h,
                       const float* __restrict__ Wd,
                       const float* __restrict__ bd,
                       float* __restrict__ out)
{
    int idx = blockIdx.x * blockDim.x + threadIdx.x;
    if (idx >= Bsz * Odim) return;
    int b = idx / Odim;
    int o = idx - b * Odim;
    float s = bd[o];
    const float* hb = h + (size_t)b * Hdim;
    for (int k = 0; k < Hdim; ++k)
        s += hb[k] * Wd[(size_t)k * Odim + o];
    out[idx] = s;
}

extern "C" void kernel_launch(void* const* d_in, const int* in_sizes, int n_in,
                              void* d_out, int out_size, void* d_ws, size_t ws_size,
                              hipStream_t stream)
{
    const float* x    = (const float*)d_in[0];
    const float* Wx   = (const float*)d_in[1];
    const float* Wh   = (const float*)d_in[2];
    const float* bias = (const float*)d_in[3];
    const float* Wd   = (const float*)d_in[4];
    const float* bd   = (const float*)d_in[5];
    float* out = (float*)d_out;

    // workspace layout (256B aligned slices)
    char* ws = (char*)d_ws;
    size_t off = 0;
    auto take = [&](size_t bytes) { void* p = ws + off; off += (bytes + 255) & ~(size_t)255; return p; };
    bf16*  WxT  = (bf16*) take((size_t)G4 * Din * sizeof(bf16));     // 2 MB
    bf16*  WhT  = (bf16*) take((size_t)G4 * Hdim * sizeof(bf16));    // 8 MB
    bf16*  hb0  = (bf16*) take((size_t)Bsz * Hdim * sizeof(bf16));   // 256 KB
    bf16*  hb1  = (bf16*) take((size_t)Bsz * Hdim * sizeof(bf16));   // 256 KB
    float* hf32 = (float*)take((size_t)Bsz * Hdim * sizeof(float));  // 512 KB
    float* c    = (float*)take((size_t)Bsz * Hdim * sizeof(float));  // 512 KB
    const size_t nX = (size_t)Bsz * Tlen * Din;
    bf16* xb = (bf16*)take(nX * sizeof(bf16));                       // 64 MB (optional)
    const bool use_xbf = (ws_size >= off);                           // runtime ws check
    (void)in_sizes; (void)n_in; (void)out_size;

    // prep: bf16 transposed weights + zeroed state (every call; deterministic)
    prep_wx<<<(G4 * Din + 255) / 256, 256, 0, stream>>>(Wx, WxT);
    prep_wh<<<(G4 * Hdim + 255) / 256, 256, 0, stream>>>(Wh, WhT);
    zero_state<<<(Bsz * Hdim + 255) / 256, 256, 0, stream>>>(hb0, c);
    if (use_xbf)
        prep_x<<<(int)((nX + 255) / 256), 256, 0, stream>>>(x, xb, (int)nX);

    // recurrent scan: one kernel per time step, bf16 h double-buffered
    bf16* hbuf[2] = {hb0, hb1};
    for (int t = 0; t < Tlen; ++t) {
        if (use_xbf)
            lstm_step<true><<<64, 128, 0, stream>>>(x, xb, WxT, WhT, bias,
                                                    hbuf[t & 1], hbuf[(t + 1) & 1],
                                                    hf32, c, t);
        else
            lstm_step<false><<<64, 128, 0, stream>>>(x, xb, WxT, WhT, bias,
                                                     hbuf[t & 1], hbuf[(t + 1) & 1],
                                                     hf32, c, t);
    }
    decode<<<(Bsz * Odim + 255) / 256, 256, 0, stream>>>(hf32, Wd, bd, out);
}
